// HTR_7464653160731
// MI455X (gfx1250) — compile-verified, matrix operand order checked by reading, not run on hardware
//
#include <hip/hip_runtime.h>
#include <hip/hip_bf16.h>

// Problem constants (from reference)
#define D_ 4
#define N_ 20000
#define F_ 128
#define R_ 64
#define E_ 320000
#define H_ 128

typedef __attribute__((ext_vector_type(16))) _Float16 v16h;
typedef __attribute__((ext_vector_type(8)))  _Float16 v8h;
typedef __attribute__((ext_vector_type(4)))  _Float16 v4h;
typedef __attribute__((ext_vector_type(8)))  float    v8f;
typedef __attribute__((ext_vector_type(4)))  float    v4f;

#if defined(__has_builtin)
#if __has_builtin(__builtin_amdgcn_global_load_async_to_lds_b128)
#define HAVE_ASYNC_LDS 1
#endif
#if __has_builtin(__builtin_amdgcn_s_wait_asynccnt)
#define HAVE_WAIT_ASYNC 1
#endif
#endif

#if defined(HAVE_ASYNC_LDS)
// Builtin signature (from hipcc diagnostic): param1 = int4 in AS(1) (global),
// param2 = int4 in AS(3) (LDS), then immediate offset + cpol.
typedef int v4i_vs __attribute__((vector_size(16)));
typedef __attribute__((address_space(1))) v4i_vs* gas_v4i;
typedef __attribute__((address_space(3))) v4i_vs* las_v4i;
#endif

__device__ inline void wait_asynccnt0() {
#if defined(HAVE_WAIT_ASYNC)
    __builtin_amdgcn_s_wait_asynccnt(0);
#else
    asm volatile("s_wait_asynccnt 0x0" ::: "memory");
#endif
}

// ---------------------------------------------------------------------------
// Fragment-packed weight region (frag = 32 lanes * 16 f16 = 1024 B, lane-major):
//   frag   0..15  : Wq    (kt 0..3 over F=128, nt 0..3 over R=64)
//   frag  16..79  : Wk[d] (16 + d*16 + kt*4 + nt)
//   frag  80..95  : mw_w1 (80 + kt*8 + nt;  kt 0..1 over R=64,  nt 0..7 over H)
//   frag  96..127 : mt_w1 (96 + kt*8 + nt;  kt 0..3 over F=128)
//   frag 128..159 : mw_w2 (128 + kt*8 + nt; kt 0..3 over H=128)
//   frag 160..191 : mt_w2 (160 + kt*8 + nt)
#define NFRAGS 192
#define PACK_BYTES (NFRAGS * 1024)

// B-operand lane layout for v_wmma_*_16x16x32 (16-bit):
//   lane l<16 : n = l,    k = {kb..kb+7, kb+16..kb+23}, kb = 0
//   lane l>=16: n = l-16, same with kb = 8
__global__ void pack_weights(const float* __restrict__ Wq,
                             const float* __restrict__ Wk,
                             const float* __restrict__ mw1,
                             const float* __restrict__ mt1,
                             const float* __restrict__ mw2,
                             const float* __restrict__ mt2,
                             _Float16* __restrict__ out) {
    int t = blockIdx.x * blockDim.x + threadIdx.x;   // NFRAGS*32 threads
    int frag = t >> 5;
    int lane = t & 31;
    const float* W;
    int Nc, kt, nt;
    if (frag < 16)        { int f = frag;       W = Wq;                      Nc = 64;  kt = f >> 2; nt = f & 3; }
    else if (frag < 80)   { int f = frag - 16;  W = Wk + (f >> 4) * (F_*R_); Nc = 64;  f &= 15; kt = f >> 2; nt = f & 3; }
    else if (frag < 96)   { int f = frag - 80;  W = mw1;                     Nc = 128; kt = f >> 3; nt = f & 7; }
    else if (frag < 128)  { int f = frag - 96;  W = mt1;                     Nc = 128; kt = f >> 3; nt = f & 7; }
    else if (frag < 160)  { int f = frag - 128; W = mw2;                     Nc = 128; kt = f >> 3; nt = f & 7; }
    else                  { int f = frag - 160; W = mt2;                     Nc = 128; kt = f >> 3; nt = f & 7; }
    int n  = nt * 16 + (lane & 15);
    int kb = kt * 32 + ((lane < 16) ? 0 : 8);
    _Float16* dst = out + (size_t)frag * 512 + lane * 16;
#pragma unroll
    for (int j = 0; j < 8; ++j) dst[j]     = (_Float16)W[(kb + j) * Nc + n];
#pragma unroll
    for (int j = 0; j < 8; ++j) dst[8 + j] = (_Float16)W[(kb + 16 + j) * Nc + n];
}

__device__ inline v16h load_bfrag(const _Float16* __restrict__ packed, int frag, int lane) {
    const v8h* p = (const v8h*)(packed + (size_t)frag * 512 + lane * 16);
    v8h lo = p[0], hi = p[1];
    v16h r;
#pragma unroll
    for (int j = 0; j < 8; ++j) { r[j] = lo[j]; r[8 + j] = hi[j]; }
    return r;
}

// A fragment (16x32 f16) from an LDS row-major tile with leading dim ld.
__device__ inline v16h lds_afrag(const _Float16* __restrict__ s, int ld, int kt, int lane) {
    int m  = lane & 15;
    int kb = (lane < 16) ? 0 : 8;
    const _Float16* p = s + m * ld + kt * 32 + kb;
    v8h lo = *(const v8h*)p;
    v8h hi = *(const v8h*)(p + 16);
    v16h r;
#pragma unroll
    for (int j = 0; j < 8; ++j) { r[j] = lo[j]; r[8 + j] = hi[j]; }
    return r;
}

// ---------------------------------------------------------------------------
// Kernel 1: eq/ek projections. One wave per (d, 16-row tile); A fragments
// shared by both GEMMs. Output layout is [N][R][D] f16 so the edge gather
// reads one contiguous v4h per matrix.
__global__ __launch_bounds__(256)
void proj_kernel(const float* __restrict__ Xs, const _Float16* __restrict__ packed,
                 _Float16* __restrict__ eq, _Float16* __restrict__ ek) {
    int wave = (blockIdx.x * blockDim.x + threadIdx.x) >> 5;  // 0..4999
    int lane = threadIdx.x & 31;
    int d  = wave / (N_ / 16);
    int mt = wave % (N_ / 16);
    const float* A = Xs + (size_t)d * N_ * F_;
    int m  = mt * 16 + (lane & 15);
    int kb = (lane < 16) ? 0 : 8;
    v16h afrag[4];
#pragma unroll
    for (int kt = 0; kt < 4; ++kt) {
        const float* row = A + (size_t)m * F_ + kt * 32 + kb;
        v4f x0 = *(const v4f*)(row +  0), x1 = *(const v4f*)(row +  4);
        v4f x2 = *(const v4f*)(row + 16), x3 = *(const v4f*)(row + 20);
#pragma unroll
        for (int j = 0; j < 4; ++j) {
            afrag[kt][j]      = (_Float16)x0[j];
            afrag[kt][4 + j]  = (_Float16)x1[j];
            afrag[kt][8 + j]  = (_Float16)x2[j];
            afrag[kt][12 + j] = (_Float16)x3[j];
        }
    }
    int col    = lane & 15;
    int rowoff = (lane < 16) ? 0 : 8;
#pragma unroll
    for (int which = 0; which < 2; ++which) {
        int fbase = which == 0 ? 0 : (16 + d * 16);
        _Float16* dst = which == 0 ? eq : ek;
#pragma unroll
        for (int nt = 0; nt < 4; ++nt) {
            v8f acc = {};
#pragma unroll
            for (int kt = 0; kt < 4; ++kt) {
                v16h b = load_bfrag(packed, fbase + kt * 4 + nt, lane);
                acc = __builtin_amdgcn_wmma_f32_16x16x32_f16(
                    false, afrag[kt], false, b, (short)0, acc, false, false);
            }
#pragma unroll
            for (int i = 0; i < 8; ++i) {
                int row = mt * 16 + rowoff + i;
                // [N][R][D] layout
                dst[((size_t)row * R_ + nt * 16 + col) * D_ + d] = (_Float16)acc[i];
            }
        }
    }
}

// ---------------------------------------------------------------------------
// Kernel 2: w_ij[e,r] = sum_d eq[n_i,r,d] * ek[n_j,r,d] -- one v4h per side.
__global__ __launch_bounds__(256)
void edge_dot(const int* __restrict__ edge_index,
              const _Float16* __restrict__ eq, const _Float16* __restrict__ ek,
              _Float16* __restrict__ wij) {
    int t = blockIdx.x * blockDim.x + threadIdx.x;  // E_*64 threads
    int e = t >> 6;
    int r = t & 63;
    int nj = edge_index[e];        // edge_index[0] = n_j
    int ni = edge_index[E_ + e];   // edge_index[1] = n_i
    v4h a4 = *(const v4h*)(eq + ((size_t)ni * R_ + r) * D_);
    v4h b4 = *(const v4h*)(ek + ((size_t)nj * R_ + r) * D_);
    float s = 0.f;
#pragma unroll
    for (int d = 0; d < D_; ++d) s += (float)a4[d] * (float)b4[d];
    wij[(size_t)e * R_ + r] = (_Float16)s;
}

// ---------------------------------------------------------------------------
// Kernel 3: dt = mlp(w_ij; R->H->F) * mlp(t_ij; F->H->F).
// 64 edges (4 M-tiles) per block; each of 8 waves owns one 16-column subtile
// and reuses every B fragment across the 4 M-tiles (4x less weight traffic).
__global__ __launch_bounds__(256)
void mlp_fused(const _Float16* __restrict__ wij, const float* __restrict__ tij,
               const _Float16* __restrict__ packed,
               const float* __restrict__ mw_b1, const float* __restrict__ mw_b2,
               const float* __restrict__ mt_b1, const float* __restrict__ mt_b2,
               float* __restrict__ out) {
    __shared__ __align__(16) _Float16 sAw[64 * 64];    //  8 KB
    __shared__ __align__(16) _Float16 sAt[64 * 128];   // 16 KB
    __shared__ __align__(16) _Float16 sHw[64 * 128];   // 16 KB
    __shared__ __align__(16) _Float16 sHt[64 * 128];   // 16 KB

    int et   = blockIdx.x;        // 64-edge tile
    int tid  = threadIdx.x;       // 0..255
    int wv   = tid >> 5;          // wave id -> column subtile
    int lane = tid & 31;

    // ---- Stage w_ij tile (64x64 f16 = 8 KB): async-to-LDS when available.
#if defined(HAVE_ASYNC_LDS)
    {
        char* g = (char*)(wij + (size_t)et * 64 * 64);
        __builtin_amdgcn_global_load_async_to_lds_b128(
            (gas_v4i)(uintptr_t)(g + tid * 16),
            (las_v4i)(uintptr_t)((char*)sAw + tid * 16), 0, 0);
        __builtin_amdgcn_global_load_async_to_lds_b128(
            (gas_v4i)(uintptr_t)(g + 4096 + tid * 16),
            (las_v4i)(uintptr_t)((char*)sAw + 4096 + tid * 16), 0, 0);
    }
#else
    {
        const v8h* src = (const v8h*)(wij + (size_t)et * 64 * 64);
        ((v8h*)sAw)[tid]       = src[tid];
        ((v8h*)sAw)[tid + 256] = src[tid + 256];
    }
#endif
    // ---- Stage t_ij tile (64x128 f32 -> f16) through VGPRs (overlaps async).
    {
        const v4f* ts = (const v4f*)(tij + (size_t)et * 64 * 128);
#pragma unroll
        for (int j = 0; j < 8; ++j) {
            v4f x = ts[tid * 8 + j];
#pragma unroll
            for (int k = 0; k < 4; ++k) sAt[tid * 32 + j * 4 + k] = (_Float16)x[k];
        }
    }
#if defined(HAVE_ASYNC_LDS)
    wait_asynccnt0();
#endif
    __syncthreads();

    int col    = wv * 16 + (lane & 15);
    int rowoff = (lane < 16) ? 0 : 8;

    // ---- Layer 1, w-branch: h_w = relu(w_ij @ mw_w1 + b1), K = 64
    {
        v8f acc[4] = {};
#pragma unroll
        for (int kt = 0; kt < 2; ++kt) {
            v16h b = load_bfrag(packed, 80 + kt * 8 + wv, lane);
#pragma unroll
            for (int mt = 0; mt < 4; ++mt) {
                v16h a = lds_afrag(sAw + mt * 16 * 64, 64, kt, lane);
                acc[mt] = __builtin_amdgcn_wmma_f32_16x16x32_f16(false, a, false, b, (short)0, acc[mt], false, false);
            }
        }
        float bias = mw_b1[col];
#pragma unroll
        for (int mt = 0; mt < 4; ++mt)
#pragma unroll
            for (int i = 0; i < 8; ++i) {
                float v = acc[mt][i] + bias;
                sHw[(mt * 16 + rowoff + i) * 128 + col] = (_Float16)(v > 0.f ? v : 0.f);
            }
    }
    // ---- Layer 1, t-branch: h_t = relu(t_ij @ mt_w1 + b1), K = 128
    {
        v8f acc[4] = {};
#pragma unroll
        for (int kt = 0; kt < 4; ++kt) {
            v16h b = load_bfrag(packed, 96 + kt * 8 + wv, lane);
#pragma unroll
            for (int mt = 0; mt < 4; ++mt) {
                v16h a = lds_afrag(sAt + mt * 16 * 128, 128, kt, lane);
                acc[mt] = __builtin_amdgcn_wmma_f32_16x16x32_f16(false, a, false, b, (short)0, acc[mt], false, false);
            }
        }
        float bias = mt_b1[col];
#pragma unroll
        for (int mt = 0; mt < 4; ++mt)
#pragma unroll
            for (int i = 0; i < 8; ++i) {
                float v = acc[mt][i] + bias;
                sHt[(mt * 16 + rowoff + i) * 128 + col] = (_Float16)(v > 0.f ? v : 0.f);
            }
    }
    __syncthreads();

    // ---- Layer 2 (both branches, K = 128) + elementwise product
    v8f aw[4] = {};
#pragma unroll
    for (int kt = 0; kt < 4; ++kt) {
        v16h b = load_bfrag(packed, 128 + kt * 8 + wv, lane);
#pragma unroll
        for (int mt = 0; mt < 4; ++mt) {
            v16h a = lds_afrag(sHw + mt * 16 * 128, 128, kt, lane);
            aw[mt] = __builtin_amdgcn_wmma_f32_16x16x32_f16(false, a, false, b, (short)0, aw[mt], false, false);
        }
    }
    v8f at[4] = {};
#pragma unroll
    for (int kt = 0; kt < 4; ++kt) {
        v16h b = load_bfrag(packed, 160 + kt * 8 + wv, lane);
#pragma unroll
        for (int mt = 0; mt < 4; ++mt) {
            v16h a = lds_afrag(sHt + mt * 16 * 128, 128, kt, lane);
            at[mt] = __builtin_amdgcn_wmma_f32_16x16x32_f16(false, a, false, b, (short)0, at[mt], false, false);
        }
    }
    float bw = mw_b2[col];
    float bt = mt_b2[col];
#pragma unroll
    for (int mt = 0; mt < 4; ++mt)
#pragma unroll
        for (int i = 0; i < 8; ++i) {
            size_t row = (size_t)et * 64 + mt * 16 + rowoff + i;
            out[row * F_ + col] = (aw[mt][i] + bw) * (at[mt][i] + bt);
        }
}

// ---------------------------------------------------------------------------
extern "C" void kernel_launch(void* const* d_in, const int* in_sizes, int n_in,
                              void* d_out, int out_size, void* d_ws, size_t ws_size,
                              hipStream_t stream) {
    const float* Xs         = (const float*)d_in[0];
    const float* t_ij       = (const float*)d_in[1];
    const int*   edge_index = (const int*)d_in[2];
    const float* Wq         = (const float*)d_in[3];
    const float* Wk         = (const float*)d_in[4];
    const float* mw_w1      = (const float*)d_in[5];
    const float* mw_b1      = (const float*)d_in[6];
    const float* mw_w2      = (const float*)d_in[7];
    const float* mw_b2      = (const float*)d_in[8];
    const float* mt_w1      = (const float*)d_in[9];
    const float* mt_b1      = (const float*)d_in[10];
    const float* mt_w2      = (const float*)d_in[11];
    const float* mt_b2      = (const float*)d_in[12];

    char* ws = (char*)d_ws;
    _Float16* packed = (_Float16*)ws;
    _Float16* eq     = (_Float16*)(ws + PACK_BYTES);
    _Float16* ek     = eq + (size_t)N_ * D_ * R_;
    _Float16* wij    = ek + (size_t)N_ * D_ * R_;

    pack_weights<<<(NFRAGS * 32) / 256, 256, 0, stream>>>(Wq, Wk, mw_w1, mt_w1, mw_w2, mt_w2, packed);
    proj_kernel<<<(D_ * (N_ / 16)) / 8, 256, 0, stream>>>(Xs, packed, eq, ek);
    edge_dot<<<(E_ * 64) / 256, 256, 0, stream>>>(edge_index, eq, ek, wij);
    mlp_fused<<<E_ / 64, 256, 0, stream>>>(wij, t_ij, packed, mw_b1, mw_b2, mt_b1, mt_b2, (float*)d_out);
}
